// GraphSageLayer_78357383349035
// MI455X (gfx1250) — matrix operand order repaired and checked by reading, not compiled, not run on hardware
//
#include <hip/hip_runtime.h>
#include <hip/hip_bf16.h>

typedef __attribute__((ext_vector_type(16))) _Float16 v16h;
typedef __attribute__((ext_vector_type(8)))  float    v8f;
typedef __attribute__((ext_vector_type(4)))  int      v4i;
typedef __attribute__((ext_vector_type(4)))  float    v4f;

#define GN 20000          // nodes
#define GF 64             // in features
#define GO 64             // out features
#define KSPLIT 5          // K-dimension split for occupancy
#define KCHUNK (GN / KSPLIT)        // 4000, multiple of 32
#define ROW_TILES (GN / 16)         // 1250
#define TOTAL_WAVES (ROW_TILES * KSPLIT)

// workspace layout (bytes)
#define ACC_BYTES ((size_t)GN * GF * 4)          // 5,120,000 f32 accumulator
#define DEG_BYTES ((size_t)GN * 4)               //    80,000 f32 degree
#define XT_BYTES  ((size_t)GF * GN * 2)          // 2,560,000 f16 x transposed [F][N]
#define WT_BYTES  ((size_t)GO * 2 * GF * 2)      //    16,384 f16 W transposed [O][2F]

// pack two 0/1 ints into two f16 (1.0h = 0x3C00) in one u32
#define PACK01(p, q) ((((unsigned)(p)) | (((unsigned)(q)) << 16)) * 0x3C00u)

// ---------------------------------------------------------------------------
// Kernel 1: zero acc/deg, build xT (f16 [F][N]) and WT (f16 [O][2F])
// ---------------------------------------------------------------------------
__global__ void gs_prep(const float* __restrict__ x,
                        const float* __restrict__ weight,
                        float* __restrict__ acc,
                        float* __restrict__ deg,
                        _Float16* __restrict__ xT,
                        _Float16* __restrict__ WT) {
  int stride = gridDim.x * blockDim.x;
  for (int idx = blockIdx.x * blockDim.x + threadIdx.x; idx < GN * GF; idx += stride) {
    acc[idx] = 0.0f;
    int n = idx / GF, f = idx % GF;              // x is [N][F] row-major
    xT[(size_t)f * GN + n] = (_Float16)x[idx];
    if (idx < GN) deg[idx] = 0.0f;
    if (idx < 2 * GF * GO) {                     // weight is [2F][O]
      int k = idx / GO, o = idx % GO;
      WT[(size_t)o * (2 * GF) + k] = (_Float16)weight[idx];
    }
  }
}

// ---------------------------------------------------------------------------
// Kernel 2: acc[0:N,0:64] += A @ x  (f16 WMMA, f32 accum), deg += rowsum(A)
// One wave = one 16-row tile over one 4000-wide K chunk.
// ---------------------------------------------------------------------------
__global__ void gs_neighbor_accum(const int* __restrict__ adj,
                                  const _Float16* __restrict__ xT,
                                  float* __restrict__ acc,
                                  float* __restrict__ deg) {
  const int wave = (blockIdx.x * blockDim.x + threadIdx.x) >> 5;
  if (wave >= TOTAL_WAVES) return;               // uniform per wave
  const int lane = threadIdx.x & 31;
  const int r = lane & 15;                       // row within tile / B column
  const int h = lane >> 4;                       // lane half
  const int rowTile = wave % ROW_TILES;
  const int kc = wave / ROW_TILES;
  const int row0 = rowTile * 16;

  const int* __restrict__ arow = adj + (size_t)(row0 + r) * GN;
  const int kend = (kc + 1) * KCHUNK;

  v8f accv0 = {}, accv1 = {}, accv2 = {}, accv3 = {};
  int degp = 0;

  for (int k0 = kc * KCHUNK; k0 < kend; k0 += 32) {
    // A-tile lane layout: halves i -> K = i + 16h
    const int* ap = arow + k0 + 16 * h;
    v4i a0 = *(const v4i*)(ap);                  // K = 16h+0..3
    v4i a1 = *(const v4i*)(ap + 4);              // K = 16h+4..7
    v4i a2 = *(const v4i*)(ap + 8);              // K = 16h+8..11
    v4i a3 = *(const v4i*)(ap + 12);             // K = 16h+12..15
    __builtin_prefetch(ap + 256, 0, 0);          // ~1KB ahead on the adj stream

    degp += a0.x + a0.y + a0.z + a0.w + a1.x + a1.y + a1.z + a1.w +
            a2.x + a2.y + a2.z + a2.w + a3.x + a3.y + a3.z + a3.w;

    union { unsigned u[8]; v16h v; } A;
    A.u[0] = PACK01(a0.x, a0.y); A.u[1] = PACK01(a0.z, a0.w);
    A.u[2] = PACK01(a1.x, a1.y); A.u[3] = PACK01(a1.z, a1.w);
    A.u[4] = PACK01(a2.x, a2.y); A.u[5] = PACK01(a2.z, a2.w);
    A.u[6] = PACK01(a3.x, a3.y); A.u[7] = PACK01(a3.z, a3.w);

    // B-tile: lane holds column f = 16t + r, nodes k0+16h .. k0+16h+15 (contiguous in xT)
    const _Float16* xcol = xT + (size_t)(k0 + 16 * h);
    v16h b0 = *(const v16h*)(xcol + (size_t)(0 * 16 + r) * GN);
    v16h b1 = *(const v16h*)(xcol + (size_t)(1 * 16 + r) * GN);
    v16h b2 = *(const v16h*)(xcol + (size_t)(2 * 16 + r) * GN);
    v16h b3 = *(const v16h*)(xcol + (size_t)(3 * 16 + r) * GN);

    accv0 = __builtin_amdgcn_wmma_f32_16x16x32_f16(false, A.v, false, b0, (short)0, accv0, false, false);
    accv1 = __builtin_amdgcn_wmma_f32_16x16x32_f16(false, A.v, false, b1, (short)0, accv1, false, false);
    accv2 = __builtin_amdgcn_wmma_f32_16x16x32_f16(false, A.v, false, b2, (short)0, accv2, false, false);
    accv3 = __builtin_amdgcn_wmma_f32_16x16x32_f16(false, A.v, false, b3, (short)0, accv3, false, false);
  }

  // flush: lanes h=0 and h=1 each hold a disjoint-K partial count for row r
  atomicAdd(&deg[row0 + r], (float)degp);

  // C layout: vgpr g, lane l -> element [M = 8h+g][n = r]
  float* accb = acc + (size_t)row0 * GF;
#pragma unroll
  for (int g = 0; g < 8; ++g) {
    atomicAdd(&accb[(size_t)(8 * h + g) * GF + 0 * 16 + r], accv0[g]);
    atomicAdd(&accb[(size_t)(8 * h + g) * GF + 1 * 16 + r], accv1[g]);
    atomicAdd(&accb[(size_t)(8 * h + g) * GF + 2 * 16 + r], accv2[g]);
    atomicAdd(&accb[(size_t)(8 * h + g) * GF + 3 * 16 + r], accv3[g]);
  }
}

// ---------------------------------------------------------------------------
// Kernel 3: out = [acc/deg, x] @ W + bias   (one wave = one 16-row tile)
// ---------------------------------------------------------------------------
__global__ void gs_output(const float* __restrict__ acc,
                          const float* __restrict__ deg,
                          const float* __restrict__ x,
                          const _Float16* __restrict__ WT,
                          const float* __restrict__ bias,
                          float* __restrict__ out) {
  const int wave = (blockIdx.x * blockDim.x + threadIdx.x) >> 5;
  if (wave >= ROW_TILES) return;                 // uniform per wave
  const int lane = threadIdx.x & 31;
  const int r = lane & 15;
  const int h = lane >> 4;
  const int row0 = wave * 16;

  const float dinv = 1.0f / deg[row0 + r];
  const float* __restrict__ accrow = acc + (size_t)(row0 + r) * GF;
  const float* __restrict__ xrow   = x   + (size_t)(row0 + r) * GF;

  // A-chunks (16x32 f16 each) over K = [x1(0:64) | x(0:64)]
  v16h Ach[4];
#pragma unroll
  for (int c = 0; c < 4; ++c) {
    const float* src = (c < 2) ? (accrow + c * 32) : (xrow + (c - 2) * 32);
    const float scale = (c < 2) ? dinv : 1.0f;
    v4f f0 = *(const v4f*)(src + 16 * h);
    v4f f1 = *(const v4f*)(src + 16 * h + 4);
    v4f f2 = *(const v4f*)(src + 16 * h + 8);
    v4f f3 = *(const v4f*)(src + 16 * h + 12);
    union { _Float16 e[16]; v16h v; } H;
#pragma unroll
    for (int j = 0; j < 4; ++j) {
      H.e[j]      = (_Float16)(f0[j] * scale);
      H.e[4 + j]  = (_Float16)(f1[j] * scale);
      H.e[8 + j]  = (_Float16)(f2[j] * scale);
      H.e[12 + j] = (_Float16)(f3[j] * scale);
    }
    Ach[c] = H.v;
  }

#pragma unroll
  for (int t = 0; t < 4; ++t) {
    v8f o = {};
    // B: lane column = output feature t*16+r; WT row is 128 contiguous halves
    const _Float16* wcol = WT + (size_t)(t * 16 + r) * (2 * GF) + 16 * h;
    o = __builtin_amdgcn_wmma_f32_16x16x32_f16(false, Ach[0], false, *(const v16h*)(wcol + 0),  (short)0, o, false, false);
    o = __builtin_amdgcn_wmma_f32_16x16x32_f16(false, Ach[1], false, *(const v16h*)(wcol + 32), (short)0, o, false, false);
    o = __builtin_amdgcn_wmma_f32_16x16x32_f16(false, Ach[2], false, *(const v16h*)(wcol + 64), (short)0, o, false, false);
    o = __builtin_amdgcn_wmma_f32_16x16x32_f16(false, Ach[3], false, *(const v16h*)(wcol + 96), (short)0, o, false, false);

    const float b = bias[t * 16 + r];
    float* orow = out + (size_t)row0 * GO + t * 16 + r;
#pragma unroll
    for (int g = 0; g < 8; ++g)
      orow[(size_t)(8 * h + g) * GO] = o[g] + b;
  }
}

// ---------------------------------------------------------------------------
extern "C" void kernel_launch(void* const* d_in, const int* in_sizes, int n_in,
                              void* d_out, int out_size, void* d_ws, size_t ws_size,
                              hipStream_t stream) {
  const float* x      = (const float*)d_in[0];
  const int*   adj    = (const int*)d_in[1];
  const float* weight = (const float*)d_in[2];
  const float* bias   = (const float*)d_in[3];
  float* out = (float*)d_out;

  char* ws = (char*)d_ws;
  float*    acc = (float*)(ws);
  float*    deg = (float*)(ws + ACC_BYTES);
  _Float16* xT  = (_Float16*)(ws + ACC_BYTES + DEG_BYTES);
  _Float16* WT  = (_Float16*)(ws + ACC_BYTES + DEG_BYTES + XT_BYTES);

  // 1) zero accumulators, build f16 transposed operands
  gs_prep<<<(GN * GF + 255) / 256, 256, 0, stream>>>(x, weight, acc, deg, xT, WT);

  // 2) A @ x via f16 WMMA (f32 accumulate) + degree counting
  gs_neighbor_accum<<<(TOTAL_WAVES * 32 + 255) / 256, 256, 0, stream>>>(adj, xT, acc, deg);

  // 3) [x1, x] @ W + bias
  gs_output<<<(ROW_TILES * 32 + 255) / 256, 256, 0, stream>>>(acc, deg, x, WT, bias, out);
}
